// QuantumOnly_39565238731108
// MI455X (gfx1250) — compile-verified, matrix operand order checked by reading, not run on hardware
//
#include <hip/hip_runtime.h>

typedef __attribute__((ext_vector_type(16))) __bf16 v16bf;
typedef __attribute__((ext_vector_type(8)))  float  v8f;

// ---------------------------------------------------------------- utilities
__global__ void k_zero(float* __restrict__ p, long long n) {
  long long i = (long long)blockIdx.x * blockDim.x + threadIdx.x;
  if (i < n) p[i] = 0.0f;
}

__global__ void k_deg(const int* __restrict__ dst, float* __restrict__ deg, int E) {
  int i = blockIdx.x * blockDim.x + threadIdx.x;
  if (i < E) atomicAdd(&deg[dst[i]], 1.0f);
}

__global__ void k_dinv(const float* __restrict__ deg, float* __restrict__ dinv, int N) {
  int i = blockIdx.x * blockDim.x + threadIdx.x;
  if (i < N) dinv[i] = rsqrtf(deg[i] + 1.0f);   // +1 self loop
}

// ---------------------------------------------------------------- GEMM 1: [N,128] x [128,32] via bf16 WMMA
// A 16x32 bf16 frag: lane m=lane&15 holds row m; kh=lane>>4 selects K halves.
// elements 0..7 -> K = kb+i ; elements 8..15 -> K = kb+16+(i-8), kb = ks*32+kh*8.
// B (W1) fragments are staged once per block in LDS as ready-to-use v16bf operands.
__global__ void k_gemm1(const float* __restrict__ x, const float* __restrict__ W,
                        float* __restrict__ h, int N) {
  __shared__ __bf16 Bf[8][32][16];                 // [ks*2+t][lane][elem] = 8 KB
  const int lane = threadIdx.x & 31;
  {
    // 256 threads fill exactly the 8 fragment slots (ks=0..3, t=0..1)
    const int combo = threadIdx.x >> 5;
    const int ks = combo >> 1, t = combo & 1;
    const int kh2 = lane >> 4;
    const int n = (lane & 15) + t * 16;
    const float* wp = W + (ks * 32 + kh2 * 8) * 32 + n;   // W1 row-major [128,32]
    __bf16* dp = &Bf[combo][lane][0];
#pragma unroll
    for (int i = 0; i < 8; ++i) {
      dp[i]     = (__bf16)wp[i * 32];
      dp[i + 8] = (__bf16)wp[(i + 16) * 32];
    }
  }
  __syncthreads();
  const int wave = threadIdx.x >> 5;
  const int tile = blockIdx.x * 8 + wave;
  const int ntiles = N >> 4;                       // N is a multiple of 16
  if (tile >= ntiles) return;                      // wave-uniform exit after barrier
  const int m  = lane & 15;
  const int kh = lane >> 4;
  const long long row = (long long)tile * 16 + m;
  const float4* xrow = (const float4*)(x + row * 128);    // b128 vector loads
  v8f c0 = {}, c1 = {};
#pragma unroll
  for (int ks = 0; ks < 4; ++ks) {
    const int q4 = ks * 8 + kh * 2;                // float4 index of kb = ks*32+kh*8
    float4 f0 = xrow[q4 + 0];
    float4 f1 = xrow[q4 + 1];
    float4 f2 = xrow[q4 + 4];
    float4 f3 = xrow[q4 + 5];
    v16bf a;
    a[0]  = (__bf16)f0.x; a[1]  = (__bf16)f0.y; a[2]  = (__bf16)f0.z; a[3]  = (__bf16)f0.w;
    a[4]  = (__bf16)f1.x; a[5]  = (__bf16)f1.y; a[6]  = (__bf16)f1.z; a[7]  = (__bf16)f1.w;
    a[8]  = (__bf16)f2.x; a[9]  = (__bf16)f2.y; a[10] = (__bf16)f2.z; a[11] = (__bf16)f2.w;
    a[12] = (__bf16)f3.x; a[13] = (__bf16)f3.y; a[14] = (__bf16)f3.z; a[15] = (__bf16)f3.w;
    v16bf b0 = *(const v16bf*)&Bf[ks * 2 + 0][lane][0];   // ds_load_b128 x2
    v16bf b1 = *(const v16bf*)&Bf[ks * 2 + 1][lane][0];
    c0 = __builtin_amdgcn_wmma_f32_16x16x32_bf16(false, a, false, b0, (short)0, c0, false, false);
    c1 = __builtin_amdgcn_wmma_f32_16x16x32_bf16(false, a, false, b1, (short)0, c1, false, false);
  }
#pragma unroll
  for (int r = 0; r < 8; ++r) {                    // C/D: VGPR r -> M = r + kh*8, N = m
    const long long orow = (long long)tile * 16 + r + kh * 8;
    h[orow * 32 + m]      = c0[r];
    h[orow * 32 + 16 + m] = c1[r];
  }
}

// ---------------------------------------------------------------- GEMM 2: [N,32] x [32,16], single WMMA per row tile
__global__ void k_gemm2(const float* __restrict__ hin, const float* __restrict__ W,
                        float* __restrict__ hout, int N) {
  __shared__ __bf16 Bf[32][16];                    // one staged fragment of W2 (1 KB)
  const int lane = threadIdx.x & 31;
  if (threadIdx.x < 32) {
    const int kh2 = lane >> 4;
    const int n = lane & 15;
    const float* wp = W + (kh2 * 8) * 16 + n;      // W2 row-major [32,16]
    __bf16* dp = &Bf[lane][0];
#pragma unroll
    for (int i = 0; i < 8; ++i) {
      dp[i]     = (__bf16)wp[i * 16];
      dp[i + 8] = (__bf16)wp[(i + 16) * 16];
    }
  }
  __syncthreads();
  const int wave = threadIdx.x >> 5;
  const int tile = blockIdx.x * 8 + wave;
  const int ntiles = N >> 4;
  if (tile >= ntiles) return;
  const int m  = lane & 15;
  const int kh = lane >> 4;
  const long long row = (long long)tile * 16 + m;
  const float4* xrow = (const float4*)(hin + row * 32 + kh * 8);
  float4 f0 = xrow[0];
  float4 f1 = xrow[1];
  float4 f2 = xrow[4];
  float4 f3 = xrow[5];
  v16bf a;
  a[0]  = (__bf16)f0.x; a[1]  = (__bf16)f0.y; a[2]  = (__bf16)f0.z; a[3]  = (__bf16)f0.w;
  a[4]  = (__bf16)f1.x; a[5]  = (__bf16)f1.y; a[6]  = (__bf16)f1.z; a[7]  = (__bf16)f1.w;
  a[8]  = (__bf16)f2.x; a[9]  = (__bf16)f2.y; a[10] = (__bf16)f2.z; a[11] = (__bf16)f2.w;
  a[12] = (__bf16)f3.x; a[13] = (__bf16)f3.y; a[14] = (__bf16)f3.z; a[15] = (__bf16)f3.w;
  v16bf b = *(const v16bf*)&Bf[lane][0];
  v8f c = {};
  c = __builtin_amdgcn_wmma_f32_16x16x32_bf16(false, a, false, b, (short)0, c, false, false);
#pragma unroll
  for (int r = 0; r < 8; ++r) {
    const long long orow = (long long)tile * 16 + r + kh * 8;
    hout[orow * 16 + m] = c[r];
  }
}

// ---------------------------------------------------------------- edge scatter-add (symmetric norm)
template <int C, int LPE>  // C channels, LPE = C/4 lanes per edge
__global__ void k_edge(const int* __restrict__ src, const int* __restrict__ dst,
                       const float* __restrict__ hin, const float* __restrict__ dinv,
                       float* __restrict__ agg, int E) {
  long long t = (long long)blockIdx.x * blockDim.x + threadIdx.x;
  long long e = t / LPE;
  int part = (int)(t % LPE);
  if (e >= E) return;
  __builtin_prefetch(src + e + 2048, 0, 1);        // global_prefetch_b8
  int s = src[e], d = dst[e];
  float coef = dinv[s] * dinv[d];
  const float4 v = *(const float4*)(hin + (long long)s * C + part * 4);
  float* ap = agg + (long long)d * C + part * 4;
  atomicAdd(ap + 0, v.x * coef);
  atomicAdd(ap + 1, v.y * coef);
  atomicAdd(ap + 2, v.z * coef);
  atomicAdd(ap + 3, v.w * coef);
}

// ---------------------------------------------------------------- self-loop + bias + ReLU
template <int C>
__global__ void k_node(const float* __restrict__ agg, const float* __restrict__ hpre,
                       const float* __restrict__ dinv, const float* __restrict__ bias,
                       float* __restrict__ out, int N) {
  long long i = (long long)blockIdx.x * blockDim.x + threadIdx.x;
  if (i >= (long long)N * C) return;
  int c = (int)(i % C);
  long long v = i / C;
  float di = dinv[v];
  float t = agg[i] + hpre[i] * di * di + bias[c];
  out[i] = t > 0.0f ? t : 0.0f;
}

// ---------------------------------------------------------------- global mean pool (sum + count)
__global__ void k_pool(const float* __restrict__ h, const int* __restrict__ batch,
                       float* __restrict__ sums, float* __restrict__ cnt, int N) {
  long long i = (long long)blockIdx.x * blockDim.x + threadIdx.x;
  if (i >= (long long)N * 16) return;
  int c = (int)(i & 15);
  long long v = i >> 4;
  int g = batch[v];
  atomicAdd(&sums[(long long)g * 16 + c], h[i]);
  if (c == 0) atomicAdd(&cnt[g], 1.0f);
}

// ---------------------------------------------------------------- projection -> tanh -> scale
__global__ void k_qin(const float* __restrict__ sums, const float* __restrict__ cnt,
                      const float* __restrict__ pW, const float* __restrict__ pb,
                      const float* __restrict__ iscale, float* __restrict__ qin, int G) {
  int i = blockIdx.x * blockDim.x + threadIdx.x;
  if (i >= G * 8) return;
  int q = i & 7, g = i >> 3;
  float ic = 1.0f / fmaxf(cnt[g], 1.0f);
  float acc = pb[q];
#pragma unroll
  for (int k = 0; k < 16; ++k) acc += sums[g * 16 + k] * ic * pW[k * 8 + q];
  qin[i] = tanhf(acc) * iscale[0];
}

// ---------------------------------------------------------------- fused 8-qubit VQC + classifier
// One wave per sample. Amplitude index a = lane*8 + r (a[7:3]=lane, a[2:0]=r).
// Qubit q (wire 0 = MSB): q<5 -> lane bit (4-q) (cross-lane via shfl_xor);
// q>=5 -> r bit (7-q) (in-register).
__global__ void k_vqc(const float* __restrict__ qin, const float* __restrict__ wts,
                      const float* __restrict__ qscale,
                      const float* __restrict__ cW1, const float* __restrict__ cb1,
                      const float* __restrict__ cW2, const float* __restrict__ cb2,
                      float* __restrict__ out, int G) {
  const int wave = threadIdx.x >> 5;
  const int lane = threadIdx.x & 31;
  const int s = blockIdx.x * 8 + wave;
  if (s >= G) return;                               // wave-uniform exit

  float ar[8], ai[8];
#pragma unroll
  for (int r = 0; r < 8; ++r) { ar[r] = 0.0f; ai[r] = 0.0f; }
  if (lane == 0) ar[0] = 1.0f;

  auto ry = [&](int q, float th) {
    float sh, ch;
    __sincosf(th * 0.5f, &sh, &ch);
    if (q >= 5) {
      int rb = 1 << (7 - q);
#pragma unroll
      for (int r = 0; r < 8; ++r)
        if (!(r & rb)) {
          int r1 = r | rb;
          float a0r = ar[r], a0i = ai[r], a1r = ar[r1], a1i = ai[r1];
          ar[r]  = ch * a0r - sh * a1r;  ai[r]  = ch * a0i - sh * a1i;
          ar[r1] = sh * a0r + ch * a1r;  ai[r1] = sh * a0i + ch * a1i;
        }
    } else {
      int lb = 1 << (4 - q);
      float sgn = (lane & lb) ? sh : -sh;
#pragma unroll
      for (int r = 0; r < 8; ++r) {
        float orr = __shfl_xor(ar[r], lb);
        float ori = __shfl_xor(ai[r], lb);
        ar[r] = ch * ar[r] + sgn * orr;
        ai[r] = ch * ai[r] + sgn * ori;
      }
    }
  };

  auto rz = [&](int q, float th) {
    float sh, ch;
    __sincosf(th * 0.5f, &sh, &ch);
#pragma unroll
    for (int r = 0; r < 8; ++r) {
      int b = (q >= 5) ? ((r >> (7 - q)) & 1) : ((lane >> (4 - q)) & 1);
      float ss = b ? sh : -sh;                      // e^{+i t/2} if bit=1 else e^{-i t/2}
      float re = ar[r], im = ai[r];
      ar[r] = re * ch - im * ss;
      ai[r] = im * ch + re * ss;
    }
  };

  auto cnot = [&](int cq, int tq) {
    if (tq >= 5) {
      int tb = 1 << (7 - tq);
#pragma unroll
      for (int r = 0; r < 8; ++r)
        if (!(r & tb)) {
          int r1 = r | tb;
          int cb = (cq >= 5) ? ((r >> (7 - cq)) & 1) : ((lane >> (4 - cq)) & 1);
          if (cb) {
            float t;
            t = ar[r]; ar[r] = ar[r1]; ar[r1] = t;
            t = ai[r]; ai[r] = ai[r1]; ai[r1] = t;
          }
        }
    } else {
      int tb = 1 << (4 - tq);
#pragma unroll
      for (int r = 0; r < 8; ++r) {
        float orr = __shfl_xor(ar[r], tb);
        float ori = __shfl_xor(ai[r], tb);
        int cb = (cq >= 5) ? ((r >> (7 - cq)) & 1) : ((lane >> (4 - cq)) & 1);
        if (cb) { ar[r] = orr; ai[r] = ori; }       // control bit identical on both partners
      }
    }
  };

  // AngleEmbedding (per-sample RY)
#pragma unroll
  for (int q = 0; q < 8; ++q) ry(q, qin[(long long)s * 8 + q]);
  // 6 layers: RY,RZ per qubit, then CNOT ring
  for (int l = 0; l < 6; ++l) {
#pragma unroll
    for (int q = 0; q < 8; ++q) {
      ry(q, wts[(l * 8 + q) * 2 + 0]);
      rz(q, wts[(l * 8 + q) * 2 + 1]);
    }
#pragma unroll
    for (int i = 0; i < 8; ++i) cnot(i, (i + 1) & 7);
  }

  // PauliZ expectation values
  float z[8];
#pragma unroll
  for (int q = 0; q < 8; ++q) z[q] = 0.0f;
#pragma unroll
  for (int r = 0; r < 8; ++r) {
    float p = ar[r] * ar[r] + ai[r] * ai[r];
    int a = lane * 8 + r;
#pragma unroll
    for (int q = 0; q < 8; ++q) z[q] += ((a >> (7 - q)) & 1) ? -p : p;
  }
#pragma unroll
  for (int q = 0; q < 8; ++q)
#pragma unroll
    for (int off = 16; off >= 1; off >>= 1) z[q] += __shfl_xor(z[q], off);

  // fused classifier: relu(z*qs @ cls_W1 + b1) @ cls_W2 + b2
  float qs = qscale[0];
  float contrib = 0.0f;
  if (lane < 16) {
    float hc = cb1[lane];
#pragma unroll
    for (int q = 0; q < 8; ++q) hc += z[q] * qs * cW1[q * 16 + lane];
    hc = hc > 0.0f ? hc : 0.0f;
    contrib = hc * cW2[lane];
  }
#pragma unroll
  for (int off = 16; off >= 1; off >>= 1) contrib += __shfl_xor(contrib, off);
  if (lane == 0) out[s] = contrib + cb2[0];
}

// ---------------------------------------------------------------- host launcher
static inline int cdiv_ll(long long a, long long b) { return (int)((a + b - 1) / b); }

extern "C" void kernel_launch(void* const* d_in, const int* in_sizes, int n_in,
                              void* d_out, int out_size, void* d_ws, size_t ws_size,
                              hipStream_t stream) {
  const float* x     = (const float*)d_in[0];
  const int*   ei    = (const int*)d_in[1];
  const int*   batch = (const int*)d_in[2];
  /* d_in[3] = num_graphs (unused; G = out_size) */
  const float* W1 = (const float*)d_in[4];
  const float* b1 = (const float*)d_in[5];
  const float* W2 = (const float*)d_in[6];
  const float* b2 = (const float*)d_in[7];
  const float* pW = (const float*)d_in[8];
  const float* pb = (const float*)d_in[9];
  const float* iscale = (const float*)d_in[10];
  const float* qscale = (const float*)d_in[11];
  const float* vqcW   = (const float*)d_in[12];
  const float* cW1 = (const float*)d_in[13];
  const float* cb1 = (const float*)d_in[14];
  const float* cW2 = (const float*)d_in[15];
  const float* cb2 = (const float*)d_in[16];
  float* outp = (float*)d_out;

  const int N = in_sizes[0] / 128;
  const int E = in_sizes[1] / 2;
  const int G = out_size;

  // workspace layout (floats): [deg N][agg1 32N][sums 16G][cnt G][dinv N][qin 8G][bufA 32N]
  float* ws = (float*)d_ws;
  long long o_agg1 = (long long)N;
  long long o_sums = o_agg1 + 32LL * N;
  long long o_cnt  = o_sums + 16LL * G;
  long long o_dinv = o_cnt + G;
  long long o_qin  = o_dinv + N;
  long long o_bufA = o_qin + 8LL * G;
  float* deg  = ws;
  float* agg1 = ws + o_agg1;       // also holds h1r after k_node<32>
  float* sums = ws + o_sums;
  float* cnt  = ws + o_cnt;
  float* dinv = ws + o_dinv;
  float* qin  = ws + o_qin;
  float* h1   = ws + o_bufA;       // [0,32N): h1; later reused
  float* h2   = h1;                // [0,16N): h2 (h1 dead by then)
  float* agg2 = h1 + 16LL * N;     // [16N,32N): agg2 / h2r

  // 1) zero accumulators: deg, agg1, sums, cnt  (contiguous [0, o_dinv))
  k_zero<<<cdiv_ll(o_dinv, 256), 256, 0, stream>>>(ws, o_dinv);
  // 2) degree + dinv
  k_deg<<<cdiv_ll(E, 256), 256, 0, stream>>>(ei + E, deg, E);
  k_dinv<<<cdiv_ll(N, 256), 256, 0, stream>>>(deg, dinv, N);
  // 3) GCN layer 1
  k_gemm1<<<cdiv_ll((long long)(N / 16), 8), 256, 0, stream>>>(x, W1, h1, N);
  k_edge<32, 8><<<cdiv_ll((long long)E * 8, 256), 256, 0, stream>>>(ei, ei + E, h1, dinv, agg1, E);
  k_node<32><<<cdiv_ll((long long)N * 32, 256), 256, 0, stream>>>(agg1, h1, dinv, b1, agg1, N);
  // 4) GCN layer 2
  k_gemm2<<<cdiv_ll((long long)(N / 16), 8), 256, 0, stream>>>(agg1, W2, h2, N);
  k_zero<<<cdiv_ll(16LL * N, 256), 256, 0, stream>>>(agg2, 16LL * N);
  k_edge<16, 4><<<cdiv_ll((long long)E * 4, 256), 256, 0, stream>>>(ei, ei + E, h2, dinv, agg2, E);
  k_node<16><<<cdiv_ll((long long)N * 16, 256), 256, 0, stream>>>(agg2, h2, dinv, b2, agg2, N);
  // 5) pool + projection
  k_pool<<<cdiv_ll((long long)N * 16, 256), 256, 0, stream>>>(agg2, batch, sums, cnt, N);
  k_qin<<<cdiv_ll((long long)G * 8, 256), 256, 0, stream>>>(sums, cnt, pW, pb, iscale, qin, G);
  // 6) VQC + classifier (one wave per graph)
  k_vqc<<<cdiv_ll(G, 8), 256, 0, stream>>>(qin, vqcW, qscale, cW1, cb1, cW2, cb2, outp, G);
}
